// Attention_19181323943977
// MI455X (gfx1250) — compile-verified, matrix operand order checked by reading, not run on hardware
//
#include <hip/hip_runtime.h>
#include <hip/hip_bf16.h>

typedef __attribute__((ext_vector_type(16))) _Float16 v16h;
typedef __attribute__((ext_vector_type(8)))  _Float16 v8h;
typedef __attribute__((ext_vector_type(8)))  float    v8f;

#define D_MODEL   1024
#define NUM_HEADS 16
#define D_KEY     64
#define SEQ_T     2048
#define BATCH     2
#define QKV_N     (3 * D_MODEL)

__device__ __forceinline__ v16h cat16(v8h lo, v8h hi) {
  v16h r;
#pragma unroll
  for (int i = 0; i < 8; ++i) { r[i] = lo[i]; r[i + 8] = hi[i]; }
  return r;
}

__device__ __forceinline__ v8f wmma_f16(v16h a, v16h b, v8f c) {
  // D = A(16x32 f16) * B(32x16 f16) + C(16x16 f32)
  return __builtin_amdgcn_wmma_f32_16x16x32_f16(
      /*neg_a=*/false, a, /*neg_b=*/false, b,
      /*c_mod=*/(short)0, c, /*reuse_a=*/false, /*reuse_b=*/false);
}

// Async global->LDS 16-byte copy (CDNA5 GLOBAL_LOAD_ASYNC_TO_LDS_B128,
// tracked by ASYNCcnt). LDS destination address = low 32 bits of the
// generic shared pointer (LDS aperture offset).
__device__ __forceinline__ void async_copy_b128(const void* gsrc,
                                                void* lds_dst) {
  unsigned lds = (unsigned)(uintptr_t)lds_dst;
  asm volatile("global_load_async_to_lds_b128 %0, %1, off"
               :: "v"(lds), "v"(gsrc)
               : "memory");
}

__device__ __forceinline__ void wait_async0() {
  asm volatile("s_wait_asynccnt 0" ::: "memory");
}

// ---------------------------------------------------------------------------
// Tiled GEMM: C[M,N] = A[M,K] * B[K,N] + bias[N]
//   A: f32 or f16 row-major, B: f32 row-major, C: f16 or f32 row-major.
// Block = 256 threads = 8 waves (4 M-waves x 2 N-waves); tile 64(M) x 128(N),
// K-step 32. Requires M%64==0, N%128==0, K%32==0 (true for this problem).
// ---------------------------------------------------------------------------
template <bool A_HALF, bool OUT_HALF>
__global__ __launch_bounds__(256) void gemm_bias_kernel(
    const void* __restrict__ Ap, const float* __restrict__ B,
    const float* __restrict__ bias, void* __restrict__ Cp,
    int M, int N, int K) {
  __shared__ __align__(16) _Float16 As[64][32];   // [m][k]
  __shared__ __align__(16) _Float16 Bs[128][40];  // [n][k] (+8 pad)

  const int tid      = threadIdx.x;
  const int lane     = tid & 31;
  const int wid      = tid >> 5;
  const int wm       = wid >> 1;   // 0..3
  const int wn       = wid & 1;    // 0..1
  const int lm       = lane & 15;
  const int half_sel = lane >> 4;
  const int koff     = half_sel * 8;

  const int n0 = blockIdx.x * 128;
  const int m0 = blockIdx.y * 64;

  v8f acc[4] = {};

  const float*    Af = (const float*)Ap;
  const _Float16* Ah = (const _Float16*)Ap;

  for (int k0 = 0; k0 < K; k0 += 32) {
    __syncthreads();
    // ---- stage A tile (64x32 f16): 16 bytes per thread ----
    {
      const int r = tid >> 2;
      const int c = (tid & 3) * 8;
      if (A_HALF) {
        // pure byte copy -> async DMA into LDS (ASYNCcnt path)
        async_copy_b128(Ah + (size_t)(m0 + r) * K + k0 + c, &As[r][c]);
      } else {
        const float* src = Af + (size_t)(m0 + r) * K + k0 + c;
        float4 f0 = *(const float4*)(src);
        float4 f1 = *(const float4*)(src + 4);
        v8h h;
        h[0] = (_Float16)f0.x; h[1] = (_Float16)f0.y;
        h[2] = (_Float16)f0.z; h[3] = (_Float16)f0.w;
        h[4] = (_Float16)f1.x; h[5] = (_Float16)f1.y;
        h[6] = (_Float16)f1.z; h[7] = (_Float16)f1.w;
        *(v8h*)&As[r][c] = h;
      }
    }
    // ---- stage B tile transposed (Bs[n][k]): 16 floats per thread ----
    {
      const int kk = tid >> 3;
      const int ng = (tid & 7) * 16;
      const float* src = B + (size_t)(k0 + kk) * N + n0 + ng;
#pragma unroll
      for (int i = 0; i < 4; ++i) {
        float4 f = *(const float4*)(src + i * 4);
        Bs[ng + i * 4 + 0][kk] = (_Float16)f.x;
        Bs[ng + i * 4 + 1][kk] = (_Float16)f.y;
        Bs[ng + i * 4 + 2][kk] = (_Float16)f.z;
        Bs[ng + i * 4 + 3][kk] = (_Float16)f.w;
      }
    }
    // prefetch next A tile -> global_prefetch_b8
    if (k0 + 32 < K) {
      const char* pre = (const char*)Ap +
          ((size_t)(m0 + (tid >> 2)) * K + k0 + 32) * (A_HALF ? 2 : 4);
      __builtin_prefetch(pre, 0, 1);
    }
    if (A_HALF) wait_async0();  // own async copies done before barrier
    __syncthreads();

    // ---- compute: A fragment per wave, 4 B fragments, 4 WMMAs ----
    v16h afrag = cat16(*(const v8h*)&As[wm * 16 + lm][koff],
                       *(const v8h*)&As[wm * 16 + lm][16 + koff]);
#pragma unroll
    for (int t = 0; t < 4; ++t) {
      const int n = wn * 64 + t * 16 + lm;
      v16h bfrag = cat16(*(const v8h*)&Bs[n][koff],
                         *(const v8h*)&Bs[n][16 + koff]);
      acc[t] = wmma_f16(afrag, bfrag, acc[t]);
    }
  }

  // ---- epilogue: bias + store ----
#pragma unroll
  for (int t = 0; t < 4; ++t) {
    const int col = n0 + wn * 64 + t * 16 + lm;
    const float bv = bias[col];
#pragma unroll
    for (int j = 0; j < 8; ++j) {
      const int row = m0 + wm * 16 + j + half_sel * 8;
      const float v = acc[t][j] + bv;
      if (OUT_HALF)
        ((_Float16*)Cp)[(size_t)row * N + col] = (_Float16)v;
      else
        ((float*)Cp)[(size_t)row * N + col] = v;
    }
  }
}

// ---------------------------------------------------------------------------
// Causal flash attention: one wave per (b, h, 16-row q tile).
// qkv: [B*T][3072] f16 (Q | K | V concatenated), ctx: [B*T][1024] f16.
// Online softmax in f32; S = Q*K^T and O += P*V via WMMA f16.
// K tile staged via async global->LDS DMA; V staged transposed via VGPRs.
// ---------------------------------------------------------------------------
__global__ __launch_bounds__(32) void flash_attn_kernel(
    const _Float16* __restrict__ qkv, _Float16* __restrict__ ctx) {
  __shared__ __align__(16) _Float16 Kt[32][64];  // [k-row][d]
  __shared__ __align__(16) _Float16 Vt[64][40];  // [d][k-row] (+8 pad)
  __shared__ __align__(16) _Float16 Pt[16][32];  // [q-row][k-col]

  const int lane     = threadIdx.x;
  const int lm       = lane & 15;
  const int half_sel = lane >> 4;
  const int koff     = half_sel * 8;

  const int qt = blockIdx.x;
  const int h  = blockIdx.y;
  const int b  = blockIdx.z;
  const size_t bt0 = (size_t)b * SEQ_T;

  // Q A-fragments: rows qt*16 + lm, d split into two 32-wide chunks
  const _Float16* qrow = qkv + (bt0 + qt * 16 + lm) * QKV_N + h * 64;
  v16h qf[2];
#pragma unroll
  for (int c = 0; c < 2; ++c)
    qf[c] = cat16(*(const v8h*)(qrow + 32 * c + koff),
                  *(const v8h*)(qrow + 32 * c + 16 + koff));

  v8f O[4] = {};
  float m_r[8], l_r[8];
#pragma unroll
  for (int j = 0; j < 8; ++j) { m_r[j] = -1e30f; l_r[j] = 0.0f; }

  const int kend = qt * 16 + 16;  // causal upper bound (exclusive)
  for (int kb = 0; kb < kend; kb += 32) {
    __syncthreads();
    // stage K tile via async DMA: Kt[r][d], 32 rows x 64 dims, 16B chunks
    for (int idx = lane; idx < 256; idx += 32) {
      const int r  = idx >> 3;
      const int c8 = (idx & 7) * 8;
      async_copy_b128(qkv + (bt0 + kb + r) * QKV_N + D_MODEL + h * 64 + c8,
                      &Kt[r][c8]);
    }
    // stage V tile transposed: Vt[d][r]
    for (int idx = lane; idx < 1024; idx += 32) {
      const int r  = idx & 31;
      const int d2 = (idx >> 5) * 2;
      const _Float16* vsrc =
          qkv + (bt0 + kb + r) * QKV_N + 2 * D_MODEL + h * 64 + d2;
      Vt[d2][r]     = vsrc[0];
      Vt[d2 + 1][r] = vsrc[1];
    }
    wait_async0();
    __syncthreads();

    // S tiles: cols [kb..kb+15] (S0) and [kb+16..kb+31] (S1)
    v8f S0 = {}, S1 = {};
#pragma unroll
    for (int c = 0; c < 2; ++c) {
      v16h b0 = cat16(*(const v8h*)&Kt[lm][32 * c + koff],
                      *(const v8h*)&Kt[lm][32 * c + 16 + koff]);
      v16h b1 = cat16(*(const v8h*)&Kt[lm + 16][32 * c + koff],
                      *(const v8h*)&Kt[lm + 16][32 * c + 16 + koff]);
      S0 = wmma_f16(qf[c], b0, S0);
      S1 = wmma_f16(qf[c], b1, S1);
    }

    // scale + causal mask + online softmax (row = j + 8*half_sel,
    // the row's 16 cols live in one width-16 shuffle group)
    const float scale = 0.125f;  // 1/sqrt(64)
    const int kc0 = kb + lm;
    const int kc1 = kb + 16 + lm;
#pragma unroll
    for (int j = 0; j < 8; ++j) {
      const int qr = qt * 16 + j + half_sel * 8;
      float s0 = S0[j] * scale, s1 = S1[j] * scale;
      if (kc0 > qr) s0 = -1e30f;
      if (kc1 > qr) s1 = -1e30f;
      float mx = fmaxf(s0, s1);
#pragma unroll
      for (int msk = 8; msk > 0; msk >>= 1)
        mx = fmaxf(mx, __shfl_xor(mx, msk, 16));
      const float m_new = fmaxf(m_r[j], mx);
      const float corr  = __expf(m_r[j] - m_new);
      const float p0    = __expf(s0 - m_new);
      const float p1    = __expf(s1 - m_new);
      float rs = p0 + p1;
#pragma unroll
      for (int msk = 8; msk > 0; msk >>= 1)
        rs += __shfl_xor(rs, msk, 16);
      l_r[j] = l_r[j] * corr + rs;
      m_r[j] = m_new;
#pragma unroll
      for (int t = 0; t < 4; ++t) O[t][j] *= corr;
      Pt[j + half_sel * 8][lm]      = (_Float16)p0;
      Pt[j + half_sel * 8][16 + lm] = (_Float16)p1;
    }
    __syncthreads();

    // O += P(16x32) * V(32x64): 4 WMMAs over the 4 d-chunks of 16
    v16h pf = cat16(*(const v8h*)&Pt[lm][koff],
                    *(const v8h*)&Pt[lm][16 + koff]);
#pragma unroll
    for (int t = 0; t < 4; ++t) {
      v16h vf = cat16(*(const v8h*)&Vt[t * 16 + lm][koff],
                      *(const v8h*)&Vt[t * 16 + lm][16 + koff]);
      O[t] = wmma_f16(pf, vf, O[t]);
    }
  }

  // normalize + store ctx (f16)
#pragma unroll
  for (int t = 0; t < 4; ++t) {
#pragma unroll
    for (int j = 0; j < 8; ++j) {
      const int row   = qt * 16 + j + half_sel * 8;
      const float inv = 1.0f / l_r[j];
      ctx[(bt0 + row) * D_MODEL + h * 64 + t * 16 + lm] =
          (_Float16)(O[t][j] * inv);
    }
  }
}

extern "C" void kernel_launch(void* const* d_in, const int* in_sizes, int n_in,
                              void* d_out, int out_size, void* d_ws,
                              size_t ws_size, hipStream_t stream) {
  const float* x     = (const float*)d_in[0];  // [B,T,1024] f32
  const float* W_qkv = (const float*)d_in[1];  // [1024,3072] f32
  const float* b_qkv = (const float*)d_in[2];  // [3072] f32
  const float* W_o   = (const float*)d_in[3];  // [1024,1024] f32
  const float* b_o   = (const float*)d_in[4];  // [1024] f32

  const int M = BATCH * SEQ_T;  // 4096 rows
  _Float16* qkv = (_Float16*)d_ws;                               // 24 MB
  _Float16* ctx = (_Float16*)((char*)d_ws +
                              (size_t)M * QKV_N * sizeof(_Float16));  // +8 MB

  // 1) QKV projection: f32 in -> f16 qkv (bias fused)
  dim3 g1(QKV_N / 128, M / 64);
  gemm_bias_kernel<false, true><<<g1, 256, 0, stream>>>(
      x, W_qkv, b_qkv, qkv, M, QKV_N, D_MODEL);

  // 2) causal flash attention -> f16 ctx
  dim3 g2(SEQ_T / 16, NUM_HEADS, BATCH);
  flash_attn_kernel<<<g2, 32, 0, stream>>>(qkv, ctx);

  // 3) output projection: f16 ctx -> f32 out (bias fused)
  dim3 g3(D_MODEL / 128, M / 64);
  gemm_bias_kernel<true, false><<<g3, 256, 0, stream>>>(
      ctx, W_o, b_o, d_out, M, D_MODEL, D_MODEL);
}
